// ConvolutionalLayer_44933947850816
// MI455X (gfx1250) — matrix operand order repaired and checked by reading, not compiled, not run on hardware
//
#include <hip/hip_runtime.h>

typedef __attribute__((ext_vector_type(16))) _Float16 v16h;
typedef __attribute__((ext_vector_type(8)))  float    v8f;

#define N_NODES 50000
#define N_EDGES 1600000
#define TILES_PER_WAVE 10   // 1250 blocks * 8 waves * 10 tiles * 16 edges = 1.6M

// ---------------------------------------------------------------------------
// Kernel 1: repack w1 [64][128] and w2 [64][64] (row-major, torch [out][in])
// into WMMA B-operand layout, converted to f16.
// B operand for v_wmma_f32_16x16x32_f16: lane L holds column N = L%16 (within an
// N-block) and 16 K-values of a 32-K step: K = 8*(L/16)+j (j<8), 16+8*(L/16)+(j-8) (j>=8).
// Tile t = kstep*4 + nb; flat index = t*512 + lane*16 + j (K slot).
// ---------------------------------------------------------------------------
__global__ __launch_bounds__(256) void repack_weights(
    const float* __restrict__ w1, const float* __restrict__ w2,
    _Float16* __restrict__ w1p, _Float16* __restrict__ w2p)
{
  for (int idx = threadIdx.x; idx < 8192; idx += 256) {
    int t = idx >> 9, lane = (idx >> 4) & 31, j = idx & 15;
    int ks = t >> 2, nb = t & 3;
    w1p[idx] = (_Float16)w1[(nb * 16 + (lane & 15)) * 128 + ks * 32 + (lane >> 4) * 8 + (j & 7) + (j >> 3) * 16];
  }
  for (int idx = threadIdx.x; idx < 4096; idx += 256) {
    int t = idx >> 9, lane = (idx >> 4) & 31, j = idx & 15;
    int ks = t >> 2, nb = t & 3;
    w2p[idx] = (_Float16)w2[(nb * 16 + (lane & 15)) * 64 + ks * 32 + (lane >> 4) * 8 + (j & 7) + (j >> 3) * 16];
  }
}

// ---------------------------------------------------------------------------
// Kernel 2: nb_sum[i][:] = sum_{e: dst[e]==i} x[src[e]][:]
// dst is sorted -> one wave per node, binary search the contiguous edge
// range. No atomics, deterministic. Lane c accumulates channels c and c+32.
// ---------------------------------------------------------------------------
__global__ __launch_bounds__(256) void nbsum_kernel(
    const float* __restrict__ x, const int* __restrict__ src,
    const int* __restrict__ dst, float* __restrict__ nbs)
{
  const int wave = threadIdx.x >> 5;
  const int lane = threadIdx.x & 31;
  const int node = blockIdx.x * 8 + wave;            // 6250*8 == 50000 exact

  int lo = 0, hi = N_EDGES;
  while (lo < hi) { int m = (lo + hi) >> 1; if (dst[m] < node) lo = m + 1; else hi = m; }
  const int start = lo;
  hi = N_EDGES;
  while (lo < hi) { int m = (lo + hi) >> 1; if (dst[m] < node + 1) lo = m + 1; else hi = m; }
  const int end = lo;

  float a0 = 0.0f, a1 = 0.0f;
  for (int e = start; e < end; ++e) {
    const float* xr = x + (size_t)src[e] * 64;
    a0 += xr[lane];
    a1 += xr[lane + 32];
  }
  nbs[(size_t)node * 64 + lane] = a0;
  nbs[(size_t)node * 64 + 32 + lane] = a1;
}

// ---------------------------------------------------------------------------
// Kernel 3: per-edge MLP via WMMA. One wave per 16-edge tile, 10 tiles/wave.
// All 24 B operands (w1: 16 tiles, w2: 8 tiles) held in VGPRs for the whole
// wave (192 VGPRs) -> no LDS/ds traffic for weights, no dscnt wait chained
// to WMMA issue. LDS used only for the per-wave hidden-tile transpose
// (D layout -> A layout): 4 packed b128 stores + 32 b16 loads per tile.
// Layer 1: [16x128] @ [128x64] -> 4 ksteps x 4 ntiles = 16 WMMAs
// Layer 2: [16x64]  @ [64x64]  -> 2 ksteps x 4 ntiles =  8 WMMAs
// ---------------------------------------------------------------------------
__global__ __launch_bounds__(256) void mlp_wmma_kernel(
    const float* __restrict__ x,  const float* __restrict__ b1,
    const float* __restrict__ b2, const int* __restrict__ src,
    const int* __restrict__ dst,  const float* __restrict__ nbs,
    const _Float16* __restrict__ w1p, const _Float16* __restrict__ w2p,
    float* __restrict__ out)
{
  __shared__ __align__(16) _Float16 hbuf[8][1024];   // 16 KB, per-wave 16x64

  const int wave = threadIdx.x >> 5;
  const int lane = threadIdx.x & 31;
  const int r    = lane & 15;        // A row / B-C-D column index of this lane
  const int half = lane >> 4;        // 0: lanes 0-15, 1: lanes 16-31
  const int kb   = half * 8;         // A-layout K sub-chunk base
  _Float16* hb = hbuf[wave];

  // ---- One-time: all WMMA B operands into registers (shared across tiles).
  v16h bw1[16];
  #pragma unroll
  for (int t = 0; t < 16; ++t) {
    union { uint4 u[2]; v16h v; } bu;
    const uint4* wp = (const uint4*)(w1p + t * 512 + lane * 16);
    bu.u[0] = wp[0]; bu.u[1] = wp[1];
    bw1[t] = bu.v;
  }
  v16h bw2[8];
  #pragma unroll
  for (int t = 0; t < 8; ++t) {
    union { uint4 u[2]; v16h v; } bu;
    const uint4* wp = (const uint4*)(w2p + t * 512 + lane * 16);
    bu.u[0] = wp[0]; bu.u[1] = wp[1];
    bw2[t] = bu.v;
  }

  // ---- One-time: biases for this lane's columns.
  float bias1[4], bias2[4];
  #pragma unroll
  for (int nb = 0; nb < 4; ++nb) {
    bias1[nb] = b1[nb * 16 + r];
    bias2[nb] = b2[nb * 16 + r];
  }

  const int tile0 = (blockIdx.x * 8 + wave) * TILES_PER_WAVE;

  for (int it = 0; it < TILES_PER_WAVE; ++it) {
    const int base = (tile0 + it) * 16;
    const int e    = base + r;
    const float* xrow = x   + (size_t)src[e] * 64;
    const float* nrow = nbs + (size_t)dst[e] * 64;

    // ---- Layer 1: accumulators initialized with bias (n = nb*16 + r) ----
    v8f c1[4];
    #pragma unroll
    for (int nb = 0; nb < 4; ++nb) {
      #pragma unroll
      for (int v = 0; v < 8; ++v) c1[nb][v] = bias1[nb];
    }

    #pragma unroll
    for (int ks = 0; ks < 4; ++ks) {
      // A operand: lane holds row r, K = ks*32 + {kb..kb+7, 16+kb..16+kb+7}
      const float* rp = (ks < 2) ? (xrow + ks * 32) : (nrow + (ks - 2) * 32);
      const float4 f0 = *(const float4*)(rp + kb);
      const float4 f1 = *(const float4*)(rp + kb + 4);
      const float4 f2 = *(const float4*)(rp + kb + 16);
      const float4 f3 = *(const float4*)(rp + kb + 20);
      v16h a;
      a[0]  = (_Float16)f0.x; a[1]  = (_Float16)f0.y; a[2]  = (_Float16)f0.z; a[3]  = (_Float16)f0.w;
      a[4]  = (_Float16)f1.x; a[5]  = (_Float16)f1.y; a[6]  = (_Float16)f1.z; a[7]  = (_Float16)f1.w;
      a[8]  = (_Float16)f2.x; a[9]  = (_Float16)f2.y; a[10] = (_Float16)f2.z; a[11] = (_Float16)f2.w;
      a[12] = (_Float16)f3.x; a[13] = (_Float16)f3.y; a[14] = (_Float16)f3.z; a[15] = (_Float16)f3.w;

      #pragma unroll
      for (int nb = 0; nb < 4; ++nb) {
        c1[nb] = __builtin_amdgcn_wmma_f32_16x16x32_f16(
            false, a, false, bw1[ks * 4 + nb], (short)0, c1[nb], false, false);
      }
    }

    // ---- ReLU + f16, write hidden tile to LDS column-major hb[n][m] ----
    // D layout: lane -> column n = nb*16 + r ; VGPR v -> row m = half*8 + v.
    #pragma unroll
    for (int nb = 0; nb < 4; ++nb) {
      union { uint4 u; _Float16 h[8]; } pk;
      #pragma unroll
      for (int v = 0; v < 8; ++v) pk.h[v] = (_Float16)fmaxf(c1[nb][v], 0.0f);
      *(uint4*)(hb + (nb * 16 + r) * 16 + half * 8) = pk.u;
    }

    // ---- Layer 2 ----
    v8f c2[4];
    #pragma unroll
    for (int nb = 0; nb < 4; ++nb) {
      #pragma unroll
      for (int v = 0; v < 8; ++v) c2[nb][v] = bias2[nb];
    }

    #pragma unroll
    for (int ks = 0; ks < 2; ++ks) {
      // Re-read hidden in A-layout: lane holds row m = r, K strided in LDS.
      v16h a;
      #pragma unroll
      for (int j = 0; j < 8; ++j) {
        a[j]     = hb[(ks * 32 + kb + j) * 16 + r];
        a[8 + j] = hb[(ks * 32 + 16 + kb + j) * 16 + r];
      }
      #pragma unroll
      for (int nb = 0; nb < 4; ++nb) {
        c2[nb] = __builtin_amdgcn_wmma_f32_16x16x32_f16(
            false, a, false, bw2[ks * 4 + nb], (short)0, c2[nb], false, false);
      }
    }

    // ---- ReLU + store: D layout -> out[(base + m)*64 + n] ----
    #pragma unroll
    for (int nb = 0; nb < 4; ++nb) {
      #pragma unroll
      for (int v = 0; v < 8; ++v) {
        out[(size_t)(base + half * 8 + v) * 64 + nb * 16 + r] = fmaxf(c2[nb][v], 0.0f);
      }
    }
  }
}

// ---------------------------------------------------------------------------
extern "C" void kernel_launch(void* const* d_in, const int* in_sizes, int n_in,
                              void* d_out, int out_size, void* d_ws, size_t ws_size,
                              hipStream_t stream) {
  const float* x  = (const float*)d_in[0];
  const float* w1 = (const float*)d_in[1];
  const float* b1 = (const float*)d_in[2];
  const float* w2 = (const float*)d_in[3];
  const float* b2 = (const float*)d_in[4];
  const int*  src = (const int*)d_in[5];
  const int*  dst = (const int*)d_in[6];
  float*      out = (float*)d_out;

  char* ws = (char*)d_ws;
  _Float16* w1p = (_Float16*)ws;              // 16 KB
  _Float16* w2p = (_Float16*)(ws + 16384);    //  8 KB
  float*    nbs = (float*)(ws + 24576);       // 50000*64*4 = 12.8 MB

  repack_weights<<<1, 256, 0, stream>>>(w1, w2, w1p, w2p);
  nbsum_kernel<<<N_NODES / 8, 256, 0, stream>>>(x, src, dst, nbs);
  mlp_wmma_kernel<<<N_EDGES / (128 * TILES_PER_WAVE), 256, 0, stream>>>(
      x, b1, b2, src, dst, nbs, w1p, w2p, out);
}
